// VectorQuantizer_33904471835547
// MI455X (gfx1250) — matrix operand order repaired and checked by reading, not compiled, not run on hardware
//
#include <hip/hip_runtime.h>
#include <hip/hip_bf16.h>

// ---- types ----
typedef __attribute__((ext_vector_type(16))) __bf16 v16bf;
typedef __attribute__((ext_vector_type(8)))  __bf16 v8bf;
typedef __attribute__((ext_vector_type(4)))  __bf16 v4bf;
typedef __attribute__((ext_vector_type(8)))  float  v8f;

// ---- problem constants (B=32,T=256,D=512,K=8192) ----
#define N_ROWS 8192
#define DIM 512
#define KCODE 8192
#define DSTEPS 16             // DIM / 32 (bf16 WMMA K-step)
#define KTILE 32              // codebook rows per LDS tile (32 KB)
#define NTILES (KCODE / KTILE)
#define ROWS_PER_WG 128       // 8 waves * 16 rows
#define ROWS_PER_WAVE 16
#define TILE_BYTES (KTILE * DIM * 2)   // 32768

// ---- workspace layout (bytes) ----
#define WS_XBF   0u
#define WS_EBF   8388608u
#define WS_ESQ   16777216u
#define WS_IDX   16809984u
#define WS_ACC   16842752u

// ---- output layout (floats) ----
#define OUT_IDX  4194304u
#define OUT_LOSS 4202496u

// =====================================================================
// 1) x fp32 -> bf16
__global__ void vq_convert_x(const float* __restrict__ x, __bf16* __restrict__ xbf) {
  int i = blockIdx.x * blockDim.x + threadIdx.x;
  float4 v = ((const float4*)x)[i];
  v4bf o;
  o[0] = (__bf16)v.x; o[1] = (__bf16)v.y; o[2] = (__bf16)v.z; o[3] = (__bf16)v.w;
  *(v4bf*)(xbf + (size_t)i * 4) = o;
}

// 2) embeddings fp32 -> bf16 + e_sq
__global__ void vq_convert_e(const float* __restrict__ e, __bf16* __restrict__ ebf,
                             float* __restrict__ esq) {
  __shared__ float red[128];
  int k = blockIdx.x, t = threadIdx.x;
  float4 v = ((const float4*)(e + (size_t)k * DIM))[t];
  v4bf o;
  o[0] = (__bf16)v.x; o[1] = (__bf16)v.y; o[2] = (__bf16)v.z; o[3] = (__bf16)v.w;
  ((v4bf*)(ebf + (size_t)k * DIM))[t] = o;
  red[t] = v.x * v.x + v.y * v.y + v.z * v.z + v.w * v.w;
  __syncthreads();
  for (int off = 64; off > 0; off >>= 1) {
    if (t < off) red[t] += red[t + off];
    __syncthreads();
  }
  if (t == 0) esq[k] = red[0];
}

// 3) zero loss accumulator
__global__ void vq_zero(float* __restrict__ acc) { *acc = 0.0f; }

// =====================================================================
// async copy of one 32 KB codebook tile into LDS (8 x b128 per thread)
__device__ __forceinline__ void stage_async(unsigned ldsBase, const __bf16* __restrict__ src,
                                            int tid) {
  const uint4* g = (const uint4*)src;
#pragma unroll
  for (int i = 0; i < 8; ++i) {
    unsigned laddr = ldsBase + (unsigned)((tid + i * 256) * 16);
    unsigned long long gaddr = (unsigned long long)(uintptr_t)(g + tid + i * 256);
    asm volatile("global_load_async_to_lds_b128 %0, %1, off"
                 :: "v"(laddr), "v"(gaddr) : "memory");
  }
}

// 4) main: bf16 WMMA dots + running argmin over all K codes
__global__ void __launch_bounds__(256)
vq_argmin(const __bf16* __restrict__ xbf, const __bf16* __restrict__ ebf,
          const float* __restrict__ esq, int* __restrict__ idxOut) {
  // two 32 KB tile buffers; argmin-reduction scratch aliases buffer 0 after the loop
  __shared__ __align__(32) char ldsRaw[2 * TILE_BYTES];   // 64 KB
  __bf16* eT0 = (__bf16*)ldsRaw;
  __bf16* eT1 = (__bf16*)(ldsRaw + TILE_BYTES);

  const int tid  = threadIdx.x;
  const int lane = tid & 31;
  const int wid  = tid >> 5;
  const int col  = lane & 15;        // N column of WMMA tile / M row of A
  const int half = lane >> 4;        // K-half selector per ISA 16-bit layouts
  const int rowBase = blockIdx.x * ROWS_PER_WG + wid * ROWS_PER_WAVE;

  const unsigned ldsB0 = (unsigned)(uintptr_t)(void*)eT0;
  const unsigned ldsB1 = (unsigned)(uintptr_t)(void*)eT1;

  // ---- preload this wave's A operand (16 rows x 512) into registers ----
  // A 16x32 in this code's contiguous k order (same as B): lane<16 holds K={0..15}, lane>=16 K={16..31}
  v16bf a[DSTEPS];
  {
    const __bf16* arow = xbf + (size_t)(rowBase + col) * DIM;
#pragma unroll
    for (int ds = 0; ds < DSTEPS; ++ds) {
      v8bf lo = *(const v8bf*)(arow + ds * 32 + half * 16);
      v8bf hi = *(const v8bf*)(arow + ds * 32 + half * 16 + 8);
#pragma unroll
      for (int j = 0; j < 8; ++j) { a[ds][j] = lo[j]; a[ds][j + 8] = hi[j]; }
    }
  }

  float best[8];
  int   bidx[8];
#pragma unroll
  for (int r = 0; r < 8; ++r) { best[r] = 3.4e38f; bidx[r] = 0; }

  // prologue: async-stage tile 0
  stage_async(ldsB0, ebf, tid);
  asm volatile("s_wait_asynccnt 0x0" ::: "memory");
  __syncthreads();

  for (int kt = 0; kt < NTILES; ++kt) {
    const int kb = kt * KTILE;
    const __bf16* cur = (kt & 1) ? eT1 : eT0;
    const unsigned nxtBase = (kt & 1) ? ldsB0 : ldsB1;

    // issue async loads for next tile; they overlap the WMMA chain below
    if (kt + 1 < NTILES)
      stage_async(nxtBase, ebf + (size_t)(kb + KTILE) * DIM, tid);
    if (kt + 2 < NTILES)
      __builtin_prefetch((const char*)(ebf + (size_t)(kb + 2 * KTILE) * DIM) +
                             (size_t)tid * 128, 0, 0);  // global_prefetch_b8

#pragma unroll
    for (int kc = 0; kc < KTILE / 16; ++kc) {
      v8f c = {};
      // B 32x16 in this code's contiguous k order (same as A): lane holds column (lane&15), K = half*16 + 0..15
      const __bf16* bbase = cur + (size_t)(kc * 16 + col) * DIM + half * 16;
      // depth-2 software pipeline of B fragments through LDS
      v16bf bq[2];
      bq[0] = *(const v16bf*)(bbase);
      bq[1] = *(const v16bf*)(bbase + 32);
#pragma unroll
      for (int ds = 0; ds < DSTEPS; ++ds) {
        c = __builtin_amdgcn_wmma_f32_16x16x32_bf16(
                false, a[ds], false, bq[ds & 1], (short)0, c, false, false);
        if (ds + 2 < DSTEPS)
          bq[ds & 1] = *(const v16bf*)(bbase + (ds + 2) * 32);
      }
      const int kcol = kb + kc * 16 + col;
      const float es = esq[kcol];
#pragma unroll
      for (int r = 0; r < 8; ++r) {
        float sc = es - 2.0f * c[r];   // d^2 - x^2 (x^2 const per row)
        if (sc < best[r]) { best[r] = sc; bidx[r] = kcol; }
      }
    }

    asm volatile("s_wait_asynccnt 0x0" ::: "memory");  // next tile landed
    __syncthreads();                                   // all waves done with cur
  }

  // ---- cross-lane argmin: C layout row m = r + 8*half, col n = lane&15 ----
  float* redV = (float*)ldsRaw;                         // reuse tile buffer 0
  int*   redI = (int*)(ldsRaw + ROWS_PER_WG * 16 * 4);  // next 8 KB
#pragma unroll
  for (int r = 0; r < 8; ++r) {
    int m = r + half * 8;
    redV[(wid * 16 + m) * 16 + col] = best[r];
    redI[(wid * 16 + m) * 16 + col] = bidx[r];
  }
  __syncthreads();
  if (tid < ROWS_PER_WG) {
    float bv = 3.4e38f; int bi = 0;
#pragma unroll
    for (int c2 = 0; c2 < 16; ++c2) {
      float v = redV[tid * 16 + c2];
      if (v < bv) { bv = v; bi = redI[tid * 16 + c2]; }
    }
    idxOut[blockIdx.x * ROWS_PER_WG + tid] = bi;
  }
}

// =====================================================================
// 5) gather quantized rows, write indices (as float), partial MSE sums
__global__ void vq_gather_loss(const float* __restrict__ x, const float* __restrict__ emb,
                               const int* __restrict__ idx, float* __restrict__ out,
                               float* __restrict__ acc) {
  __shared__ float red[128];
  int row = blockIdx.x, t = threadIdx.x;
  int k = idx[row];
  float4 q  = ((const float4*)(emb + (size_t)k * DIM))[t];
  float4 xv = ((const float4*)(x + (size_t)row * DIM))[t];
  ((float4*)(out + (size_t)row * DIM))[t] = q;     // quantized_st == quantized
  float d0 = xv.x - q.x, d1 = xv.y - q.y, d2 = xv.z - q.z, d3 = xv.w - q.w;
  red[t] = d0 * d0 + d1 * d1 + d2 * d2 + d3 * d3;
  __syncthreads();
  for (int off = 64; off > 0; off >>= 1) {
    if (t < off) red[t] += red[t + off];
    __syncthreads();
  }
  if (t == 0) {
    atomicAdd(acc, red[0]);
    out[(size_t)OUT_IDX + row] = (float)k;
  }
}

// 6) finalize losses
__global__ void vq_finalize(const float* __restrict__ acc, float* __restrict__ out) {
  float mse = *acc / (float)((size_t)N_ROWS * DIM);
  out[OUT_LOSS + 0] = 0.25f * mse;   // commitment_loss
  out[OUT_LOSS + 1] = mse;           // codebook_loss
  out[OUT_LOSS + 2] = 1.25f * mse;   // total_loss
}

// =====================================================================
extern "C" void kernel_launch(void* const* d_in, const int* in_sizes, int n_in,
                              void* d_out, int out_size, void* d_ws, size_t ws_size,
                              hipStream_t stream) {
  (void)in_sizes; (void)n_in; (void)out_size; (void)ws_size;
  const float* x   = (const float*)d_in[0];
  const float* emb = (const float*)d_in[1];
  char* ws = (char*)d_ws;
  __bf16* xbf = (__bf16*)(ws + WS_XBF);
  __bf16* ebf = (__bf16*)(ws + WS_EBF);
  float*  esq = (float*)(ws + WS_ESQ);
  int*    idx = (int*)(ws + WS_IDX);
  float*  acc = (float*)(ws + WS_ACC);
  float*  out = (float*)d_out;

  vq_convert_x<<<(N_ROWS * DIM) / 4 / 256, 256, 0, stream>>>(x, xbf);
  vq_convert_e<<<KCODE, 128, 0, stream>>>(emb, ebf, esq);
  vq_zero<<<1, 1, 0, stream>>>(acc);
  vq_argmin<<<N_ROWS / ROWS_PER_WG, 256, 0, stream>>>(xbf, ebf, esq, idx);
  vq_gather_loss<<<N_ROWS, 128, 0, stream>>>(x, emb, idx, out, acc);
  vq_finalize<<<1, 1, 0, stream>>>(acc, out);
}
